// Top1Router_26611617366083
// MI455X (gfx1250) — matrix operand order repaired and checked by reading, not compiled, not run on hardware
//
#include <hip/hip_runtime.h>
#include <hip/hip_bf16.h>

typedef __attribute__((ext_vector_type(16))) _Float16 v16h;
typedef __attribute__((ext_vector_type(8)))  float    v8f;
typedef __attribute__((ext_vector_type(4)))  float    v4f;

#define S_TOK   8192
#define E_EXP   64
#define CAP     256
#define CHUNK   256          // tokens per rank-block
#define NCHUNK  (S_TOK / CHUNK)   // 32
#define PLANE   (E_EXP * CAP)     // 16384 floats per token-plane
#define TOTAL   ((size_t)S_TOK * PLANE)

// ---------------------------------------------------------------------------
// Kernel 1: per-token argmax + softmax value at the argmax.
// softmax(row)[argmax] = exp(0)/sum(exp(x-max)) = 1/sum(exp(x-max)).
// 2 MB of reads total -> negligible; keep it register-resident.
// ---------------------------------------------------------------------------
__global__ void route_kernel(const float* __restrict__ in,
                             int* __restrict__ idx_out,
                             float* __restrict__ p_out) {
    int t = blockIdx.x * blockDim.x + threadIdx.x;
    if (t >= S_TOK) return;
    const v4f* r4 = (const v4f*)(in + (size_t)t * E_EXP);
    v4f buf[16];
#pragma unroll
    for (int k = 0; k < 16; ++k) buf[k] = r4[k];

    float m = -3.4e38f;
    int   am = 0;
#pragma unroll
    for (int k = 0; k < 16; ++k)
#pragma unroll
        for (int c = 0; c < 4; ++c) {
            float x = buf[k][c];
            int   i = k * 4 + c;
            if (x > m) { m = x; am = i; }   // first-max wins (strict >)
        }
    float sum = 0.f;
#pragma unroll
    for (int k = 0; k < 16; ++k)
#pragma unroll
        for (int c = 0; c < 4; ++c) sum += __expf(buf[k][c] - m);

    idx_out[t] = am;
    p_out[t]   = 1.0f / sum;
}

// ---------------------------------------------------------------------------
// Kernel 2: order-dependent exclusive cumsum (moe_cumsum) via WMMA.
// Each wave owns 32 tokens. ranks = L_strict(32x32) @ onehot(32x64),
// decomposed into 2 x 4 v_wmma_f32_16x16x32_f16:
//   D_lo = [L16 | 0  ] * B   -> exclusive ranks of tokens 0..15
//   D_hi = [1s  | L16] * B   -> exclusive ranks of tokens 16..31
// per 16-expert group g (B[k,n] = idx[tok k]==16g+n).
// Group/block totals are order-independent -> LDS histogram + tiny scans.
// ---------------------------------------------------------------------------
__global__ void rank_kernel(const int* __restrict__ idx,
                            int* __restrict__ intra,
                            int* __restrict__ hist) {
    __shared__ int   s_idx[CHUNK];
    __shared__ int   s_ghist[8][E_EXP];
    __shared__ int   s_gpref[8][E_EXP];
    __shared__ float s_rk[8][32][16];

    const int tid  = threadIdx.x;
    const int lane = tid & 31;
    const int w    = tid >> 5;              // wave id = 32-token group id
    const int t    = blockIdx.x * CHUNK + tid;

    ((int*)s_ghist)[tid]       = 0;
    ((int*)s_ghist)[tid + 256] = 0;
    __syncthreads();

    const int myidx = idx[t];
    s_idx[tid] = myidx;
    atomicAdd(&s_ghist[w][myidx], 1);
    __syncthreads();

    // exclusive scan of the 8 group histograms per expert + chunk totals
    if (tid < E_EXP) {
        int run = 0;
#pragma unroll
        for (int g = 0; g < 8; ++g) { s_gpref[g][tid] = run; run += s_ghist[g][tid]; }
        hist[blockIdx.x * E_EXP + tid] = run;
    }
    __syncthreads();

    // --- constant A tiles (ISA 16-bit A layout: lane<16 holds K{0..7,16..23},
    //     lane>=16 holds K{8..15,24..31}; element h -> K = (h<8?h:h+8)+base) ---
    const int M  = lane & 15;
    const int ka = (lane >> 4) << 3;        // 0 or 8
    v16h a_lo, a_hi;
#pragma unroll
    for (int h = 0; h < 16; ++h) {
        int K = ((h < 8) ? h : h + 8) + ka;
        a_lo[h] = (K < M)      ? (_Float16)1.0f : (_Float16)0.0f;
        a_hi[h] = (K < M + 16) ? (_Float16)1.0f : (_Float16)0.0f;
    }

    // --- B layout: lane n%16 = column, K = (h<8?h:h+8) + (lane>=16 ? 8 : 0) ---
    const int nb = lane & 15;
    const int kb = (lane >> 4) << 4;        // 0 or 16
    const int tokbase = w * 32;
    float myrank = 0.f;

    for (int g = 0; g < 4; ++g) {           // uniform loop: EXEC all-1 at WMMA
        v16h b;
#pragma unroll
        for (int h = 0; h < 16; ++h)
            b[h] = (s_idx[tokbase + ka + ((h < 8) ? h : h + 8)] == g * 16 + nb)
                       ? (_Float16)1.0f : (_Float16)0.0f;

        v8f c0 = {}; v8f c1 = {};
        c0 = __builtin_amdgcn_wmma_f32_16x16x32_f16(false, a_lo, false, b,
                                                    (short)0, c0, false, false);
        c1 = __builtin_amdgcn_wmma_f32_16x16x32_f16(false, a_hi, false, b,
                                                    (short)0, c1, false, false);

        // D layout: lane<16 rows v, lane>=16 rows v+8; column = lane&15
        const int mrow = (lane >> 4) << 3;
#pragma unroll
        for (int v = 0; v < 8; ++v) {
            s_rk[w][mrow + v][nb]      = c0[v];
            s_rk[w][16 + mrow + v][nb] = c1[v];
        }
        __syncthreads();
        if ((myidx >> 4) == g) myrank = s_rk[w][lane][myidx & 15];
        __syncthreads();
    }

    intra[t] = (int)(myrank + 0.5f) + s_gpref[w][myidx];
}

// ---------------------------------------------------------------------------
// Kernel 3: exclusive scan of per-chunk histograms (32 chunks x 64 experts).
// ---------------------------------------------------------------------------
__global__ void scan_kernel(const int* __restrict__ hist,
                            int* __restrict__ offs) {
    int e = threadIdx.x;
    if (e < E_EXP) {
        int run = 0;
#pragma unroll
        for (int c = 0; c < NCHUNK; ++c) {
            offs[c * E_EXP + e] = run;
            run += hist[c * E_EXP + e];
        }
    }
}

// ---------------------------------------------------------------------------
// Kernel 4: the 1 GiB streaming writer (46 us HBM floor). One block per
// token-plane; 32 NT float4 stores per thread across both output tensors.
// The single nonzero (combine=p, sec=1.0) is folded into whichever float4
// the owning thread writes -> no ordering hazard, fully deterministic.
// ---------------------------------------------------------------------------
__global__ void emit_kernel(const int* __restrict__ idx,
                            const float* __restrict__ pv,
                            const int* __restrict__ intra,
                            const int* __restrict__ offs,
                            float* __restrict__ out) {
    const int t   = blockIdx.x;
    const int tid = threadIdx.x;
    const int e   = idx[t];
    const int r   = intra[t] + offs[(t >> 8) * E_EXP + e];
    const bool kept = (r < CAP);
    const float p = pv[t];
    const int pos  = e * CAP + r;       // < 16384 when kept
    const int jpos = pos >> 2;
    const int comp = pos & 3;

    float* base_c = out + (size_t)t * PLANE;
    float* base_s = out + TOTAL + (size_t)t * PLANE;

#pragma unroll
    for (int k = 0; k < 16; ++k) {
        int j = k * 256 + tid;          // 4096 float4s / plane, coalesced
        v4f v = {0.f, 0.f, 0.f, 0.f};
        if (kept && j == jpos) v[comp] = p;
        __builtin_nontemporal_store(v, (v4f*)base_c + j);
    }
#pragma unroll
    for (int k = 0; k < 16; ++k) {
        int j = k * 256 + tid;
        v4f v = {0.f, 0.f, 0.f, 0.f};
        if (kept && j == jpos) v[comp] = 1.0f;   // sec_mask as float
        __builtin_nontemporal_store(v, (v4f*)base_s + j);
    }
}

// ---------------------------------------------------------------------------
extern "C" void kernel_launch(void* const* d_in, const int* in_sizes, int n_in,
                              void* d_out, int out_size, void* d_ws, size_t ws_size,
                              hipStream_t stream) {
    const float* in = (const float*)d_in[0];
    float* out = (float*)d_out;

    // workspace layout (~112 KB)
    int*   w_idx   = (int*)d_ws;                    // S_TOK
    float* w_p     = (float*)(w_idx + S_TOK);       // S_TOK
    int*   w_intra = (int*)(w_p + S_TOK);           // S_TOK
    int*   w_hist  = w_intra + S_TOK;               // NCHUNK*E_EXP
    int*   w_offs  = w_hist + NCHUNK * E_EXP;       // NCHUNK*E_EXP

    route_kernel<<<S_TOK / 256, 256, 0, stream>>>(in, w_idx, w_p);
    rank_kernel <<<NCHUNK, CHUNK, 0, stream>>>(w_idx, w_intra, w_hist);
    scan_kernel <<<1, 64, 0, stream>>>(w_hist, w_offs);
    emit_kernel <<<S_TOK, 256, 0, stream>>>(w_idx, w_p, w_intra, w_offs, out);
}
